// CAM_77318001262619
// MI455X (gfx1250) — matrix operand order repaired and checked by reading, not compiled, not run on hardware
//
#include <hip/hip_runtime.h>

// ---------------------------------------------------------------------------
// CAM (channel attention) for inputs [16, 64, 64, 512] fp32, gamma [1] fp32.
//   G = A^T A  (per batch, A = [4096, 512])        -> bf16 WMMA, fp32 acc
//   S = softmax(G, axis=-1)                        -> fp32, in-place in ws
//   out = gamma * (A S) + input                    -> bf16 WMMA, fp32 acc
// ---------------------------------------------------------------------------

typedef __bf16 bf16_t;
typedef bf16_t v16bf __attribute__((ext_vector_type(16)));
typedef bf16_t v8bf  __attribute__((ext_vector_type(8)));
typedef float  v8f   __attribute__((ext_vector_type(8)));
typedef float  v4f   __attribute__((ext_vector_type(4)));

#define BATCH 16
#define NPIX  4096   // 64*64
#define CH    512
#define KCH   64     // K-chunk staged through LDS (2 WMMA substeps)
#define LDK   72     // padded K row length (144 bytes, multiple of 16)

__device__ __forceinline__ bf16_t f2bf(float f) {
    unsigned u = __builtin_bit_cast(unsigned, f);
    u += 0x7FFFu + ((u >> 16) & 1u);          // round-to-nearest-even
    return __builtin_bit_cast(bf16_t, (unsigned short)(u >> 16));
}

__device__ __forceinline__ unsigned pack2(float a, float b) {
    return (unsigned)__builtin_bit_cast(unsigned short, f2bf(a))
         | ((unsigned)__builtin_bit_cast(unsigned short, f2bf(b)) << 16);
}

// Operand A (16x32 bf16, M x K): lane m = lane%16, halves interleaved in
// 8-wide K groups by lane half:  elems[0..7]  = K = kh*8 + 0..7,
//                                elems[8..15] = K = 16 + kh*8 + 0..7.
__device__ __forceinline__ v16bf load_opA(const bf16_t* row, int kh) {
    v8bf lo = *(const v8bf*)(row + kh * 8);
    v8bf hi = *(const v8bf*)(row + 16 + kh * 8);
    return __builtin_shufflevector(lo, hi, 0,1,2,3,4,5,6,7,8,9,10,11,12,13,14,15);
}

// Operand B (32x16 bf16, K x N): lane n = lane%16, elems[0..7] = K = kh*8 + 0..7, elems[8..15] = K = 16 + kh*8 + 0..7.
__device__ __forceinline__ v16bf load_opB(const bf16_t* col, int kh) {
    v8bf lo = *(const v8bf*)(col + kh * 8);
    v8bf hi = *(const v8bf*)(col + 16 + kh * 8);
    return __builtin_shufflevector(lo, hi, 0,1,2,3,4,5,6,7,8,9,10,11,12,13,14,15);
}

// ---------------------------------------------------------------------------
// Kernel 1: G[b] = A^T A, per-block 128x128 tile of G.
// grid = (16 tiles, 16 batches), block = 256 (8 waves).
// ---------------------------------------------------------------------------
__global__ __launch_bounds__(256) void cam_gram(const float* __restrict__ A,
                                                float* __restrict__ G) {
    const int b   = blockIdx.y;
    const int ct0 = (blockIdx.x >> 2) * 128;   // G row range (c)
    const int dt0 = (blockIdx.x & 3)  * 128;   // G col range (d)
    const float* Ab = A + (size_t)b * NPIX * CH;

    __shared__ alignas(16) bf16_t ldsR[128][LDK];  // A^T rows: [c][n_local]
    __shared__ alignas(16) bf16_t ldsC[128][LDK];  // A   cols: [d][n_local]

    const int tid  = threadIdx.x;
    const int lane = tid & 31;
    const int wave = tid >> 5;
    const int m    = lane & 15;
    const int kh   = lane >> 4;

    v8f acc[8] = {};

    for (int n0 = 0; n0 < NPIX; n0 += KCH) {
        __syncthreads();
        #pragma unroll
        for (int i = 0; i < 8; ++i) {
            int lin = tid + i * 256;             // 64 rows x 32 float4
            int r   = lin >> 5;                  // n_local 0..63
            int c4  = (lin & 31) << 2;           // channel offset 0..124
            v4f vr = *(const v4f*)&Ab[(size_t)(n0 + r) * CH + ct0 + c4];
            v4f vc = *(const v4f*)&Ab[(size_t)(n0 + r) * CH + dt0 + c4];
            #pragma unroll
            for (int j = 0; j < 4; ++j) {
                ldsR[c4 + j][r] = f2bf(vr[j]);
                ldsC[c4 + j][r] = f2bf(vc[j]);
            }
        }
        __syncthreads();

        #pragma unroll
        for (int ks = 0; ks < 2; ++ks) {          // two K=32 substeps
            const int kb = ks * 32;
            v16bf aop = load_opA(&ldsR[wave * 16 + m][kb], kh);
            v16bf bops[8];
            #pragma unroll
            for (int t = 0; t < 8; ++t)
                bops[t] = load_opB(&ldsC[t * 16 + m][kb], kh);
            #pragma unroll
            for (int t = 0; t < 8; ++t)           // back-to-back WMMA burst
                acc[t] = __builtin_amdgcn_wmma_f32_16x16x32_bf16(
                    false, aop, false, bops[t], (short)0, acc[t], false, false);
        }
    }

    // C/D layout: lane col = lane%16; VGPR r -> row r (+8 for upper lane half).
    const int col   = lane & 15;
    const int rbase = (lane >> 4) << 3;
    #pragma unroll
    for (int t = 0; t < 8; ++t) {
        #pragma unroll
        for (int r = 0; r < 8; ++r) {
            size_t idx = ((size_t)b * CH + ct0 + wave * 16 + rbase + r) * CH
                       + dt0 + t * 16 + col;
            G[idx] = acc[t][r];
        }
    }
}

// ---------------------------------------------------------------------------
// Kernel 2: in-place row softmax over last axis of G  (8192 rows of 512).
// ---------------------------------------------------------------------------
__global__ __launch_bounds__(256) void cam_softmax(float* __restrict__ G) {
    float* p = G + (size_t)blockIdx.x * CH;
    const int tid  = threadIdx.x;
    const int lane = tid & 31;
    const int wave = tid >> 5;
    __shared__ float red[8];

    float v0 = p[tid], v1 = p[tid + 256];

    float mx = fmaxf(v0, v1);
    #pragma unroll
    for (int off = 16; off > 0; off >>= 1)
        mx = fmaxf(mx, __shfl_xor(mx, off, 32));
    if (lane == 0) red[wave] = mx;
    __syncthreads();
    mx = red[0];
    #pragma unroll
    for (int w = 1; w < 8; ++w) mx = fmaxf(mx, red[w]);
    __syncthreads();

    float e0 = __expf(v0 - mx), e1 = __expf(v1 - mx);
    float s = e0 + e1;
    #pragma unroll
    for (int off = 16; off > 0; off >>= 1)
        s += __shfl_xor(s, off, 32);
    if (lane == 0) red[wave] = s;
    __syncthreads();
    s = red[0];
    #pragma unroll
    for (int w = 1; w < 8; ++w) s += red[w];

    float inv = 1.0f / s;
    p[tid]       = e0 * inv;
    p[tid + 256] = e1 * inv;
}

// ---------------------------------------------------------------------------
// Kernel 3: out = gamma * (A @ S) + input.  Per-block 128(n) x 128(d) tile.
// grid = (32, 4, 16), block = 256 (8 waves).
// ---------------------------------------------------------------------------
__global__ __launch_bounds__(256) void cam_attend(const float* __restrict__ A,
                                                  const float* __restrict__ S,
                                                  const float* __restrict__ gamma,
                                                  float* __restrict__ out) {
    const int b  = blockIdx.z;
    const int n0 = blockIdx.x * 128;
    const int d0 = blockIdx.y * 128;
    const float* Ab = A + (size_t)b * NPIX * CH;
    const float* Sb = S + (size_t)b * CH * CH;

    __shared__ alignas(16) bf16_t ldsA[128][LDK];  // [n_local][k_local]
    __shared__ alignas(16) bf16_t ldsB[128][LDK];  // [d_local][k_local] (S^T tile)

    const int tid  = threadIdx.x;
    const int lane = tid & 31;
    const int wave = tid >> 5;
    const int m    = lane & 15;
    const int kh   = lane >> 4;

    v8f acc[8] = {};

    for (int k0 = 0; k0 < CH; k0 += KCH) {
        __syncthreads();
        #pragma unroll
        for (int i = 0; i < 8; ++i) {
            int lin = tid + i * 256;
            // A tile: 128 rows x 16 float4 (K contiguous in mem and LDS -> b64 pack)
            int an = lin >> 4;
            int ak = (lin & 15) << 2;
            v4f va = *(const v4f*)&Ab[(size_t)(n0 + an) * CH + k0 + ak];
            *(uint2*)&ldsA[an][ak] = make_uint2(pack2(va[0], va[1]),
                                                pack2(va[2], va[3]));
            // S tile: 64 k-rows x 32 float4 of d; store transposed
            int sk = lin >> 5;
            int sd = (lin & 31) << 2;
            v4f vs = *(const v4f*)&Sb[(size_t)(k0 + sk) * CH + d0 + sd];
            #pragma unroll
            for (int j = 0; j < 4; ++j)
                ldsB[sd + j][sk] = f2bf(vs[j]);
        }
        __syncthreads();

        #pragma unroll
        for (int ks = 0; ks < 2; ++ks) {
            const int kb = ks * 32;
            v16bf aop = load_opA(&ldsA[wave * 16 + m][kb], kh);
            v16bf bops[8];
            #pragma unroll
            for (int t = 0; t < 8; ++t)
                bops[t] = load_opB(&ldsB[t * 16 + m][kb], kh);
            #pragma unroll
            for (int t = 0; t < 8; ++t)
                acc[t] = __builtin_amdgcn_wmma_f32_16x16x32_bf16(
                    false, aop, false, bops[t], (short)0, acc[t], false, false);
        }
    }

    const float g   = gamma[0];
    const int col   = lane & 15;
    const int rbase = (lane >> 4) << 3;
    #pragma unroll
    for (int t = 0; t < 8; ++t) {
        #pragma unroll
        for (int r = 0; r < 8; ++r) {
            size_t idx = ((size_t)b * NPIX + n0 + wave * 16 + rbase + r) * CH
                       + d0 + t * 16 + col;
            out[idx] = fmaf(g, acc[t][r], A[idx]);   // A == original input
        }
    }
}

// ---------------------------------------------------------------------------
extern "C" void kernel_launch(void* const* d_in, const int* in_sizes, int n_in,
                              void* d_out, int out_size, void* d_ws, size_t ws_size,
                              hipStream_t stream) {
    const float* inp   = (const float*)d_in[0];
    const float* gamma = (const float*)d_in[1];
    float*       out   = (float*)d_out;
    float*       G     = (float*)d_ws;   // 16*512*512 fp32 = 16.78 MB

    cam_gram<<<dim3(16, BATCH), 256, 0, stream>>>(inp, G);
    cam_softmax<<<dim3(BATCH * CH), 256, 0, stream>>>(G);
    cam_attend<<<dim3(NPIX / 128, CH / 128, BATCH), 256, 0, stream>>>(inp, G, gamma, out);
}
